// RWKV_TimeMix_47897475285605
// MI455X (gfx1250) — matrix operand order repaired and checked by reading, not compiled, not run on hardware
//
#include <hip/hip_runtime.h>

// ---------------------------------------------------------------------------
// RWKV6 TimeMix forward for MI455X (gfx1250, wave32, WMMA).
// GEMM-dominated (~88 GFLOP) -> v_wmma_f32_16x16x32_bf16, f32 accumulate.
// Weights are transpose-packed ONCE per launch into the LDS-ready bf16-pair
// layout; activations feeding GEMMs are produced pre-packed. GEMM inner loop:
// GLOBAL_LOAD_ASYNC_TO_LDS_B128 tile copies (ASYNCcnt) when available, else
// uint4 global loads -> ds_store_b128; ping-pong LDS, 1 barrier per 32-K step,
// ds_load_b128 fragments -> 8 WMMAs/wave.
// ---------------------------------------------------------------------------

#define DEV_INLINE __device__ __forceinline__

typedef __attribute__((ext_vector_type(16))) __bf16    v16bf;
typedef __attribute__((ext_vector_type(8)))  float     v8f;
typedef __attribute__((ext_vector_type(4)))  unsigned  v4u;
typedef __attribute__((__vector_size__(16))) int       v4i;   // builtin's pointee type

#define GLOBAL_AS __attribute__((address_space(1)))
#define LDS_AS    __attribute__((address_space(3)))

#if __has_builtin(__builtin_amdgcn_global_load_async_to_lds_b128)
#define USE_ASYNC_LDS 1
#else
#define USE_ASYNC_LDS 0
#endif

constexpr int Bb = 2, Tt = 1024, Cc = 2048, Hh = 32;
constexpr int BT = Bb * Tt;                 // 2048 rows
constexpr float GN_EPS = 1e-5f * 64.0f;     // reference: 1e-5 * 8**2

DEV_INLINE unsigned pack_bf16(float a, float b) {
  unsigned ua = __builtin_bit_cast(unsigned, a);
  unsigned ub = __builtin_bit_cast(unsigned, b);
  ua = (ua + 0x7FFFu + ((ua >> 16) & 1u)) >> 16;   // round-to-nearest-even
  ub = (ub + 0x7FFFu + ((ub >> 16) & 1u)) >> 16;
  return (ua & 0xFFFFu) | (ub << 16);
}

DEV_INLINE v4u zero4() { v4u z; z.x = 0; z.y = 0; z.z = 0; z.w = 0; return z; }

DEV_INLINE void async_wait0() {
#if __has_builtin(__builtin_amdgcn_s_wait_asynccnt)
  __builtin_amdgcn_s_wait_asynccnt(0);
#else
  asm volatile("s_wait_asynccnt 0" ::: "memory");
#endif
}

// ---------------------------------------------------------------------------
// Transpose-pack a weight W[K,N] (f32, row-major) into Wt[N][K/2] where each
// dword packs bf16(W[2p,n]) | bf16(W[2p+1,n])<<16 -- the exact pair layout the
// GEMM's B-operand LDS tiles want. LDS-tiled so both sides stay coalesced.
// ---------------------------------------------------------------------------
__global__ __launch_bounds__(256) void packT_kernel(
    const float* __restrict__ W, unsigned* __restrict__ Wt, int K, int N) {
  __shared__ float tile[64][65];
  const int kb = blockIdx.y * 64, nb = blockIdx.x * 64;
  const int tid = threadIdx.x;
#pragma unroll
  for (int it = 0; it < 16; ++it) {
    int e = it * 256 + tid;
    int kk = e >> 6, n = e & 63;
    float v = 0.f;
    if (nb + n < N) v = W[(long)(kb + kk) * N + nb + n];
    tile[kk][n] = v;
  }
  __syncthreads();
#pragma unroll
  for (int it = 0; it < 8; ++it) {
    int e = it * 256 + tid;
    int n = e >> 5, p = e & 31;
    if (nb + n < N)
      Wt[(long)(nb + n) * (K >> 1) + (kb >> 1) + p] =
          pack_bf16(tile[2 * p][n], tile[2 * p + 1][n]);
  }
}

// ---------------------------------------------------------------------------
// Token shift (pair-per-thread): xx = shift(x)-x (f32, for mix); packed
// bf16 xxx = x + xx*maa_x (GEMM A operand).
// ---------------------------------------------------------------------------
__global__ __launch_bounds__(256) void prep_kernel(
    const float* __restrict__ x, const float* __restrict__ maa_x,
    float* __restrict__ xx, unsigned* __restrict__ xxx_pk) {
  long gid = (long)blockIdx.x * 256 + threadIdx.x;   // over BT*C/2
  int c2 = (int)(gid % (Cc / 2));
  long bt = gid / (Cc / 2);
  int t = (int)(bt % Tt);
  long base = bt * Cc + 2 * c2;
  float2 xv = *(const float2*)(x + base);
  float2 pv;
  pv.x = 0.f; pv.y = 0.f;
  if (t != 0) pv = *(const float2*)(x + base - Cc);
  float d0 = pv.x - xv.x, d1 = pv.y - xv.y;
  float2 dd; dd.x = d0; dd.y = d1;
  *(float2*)(xx + base) = dd;
  float2 mx = *(const float2*)(maa_x + 2 * c2);
  xxx_pk[gid] = pack_bf16(xv.x + d0 * mx.x, xv.y + d1 * mx.y);
}

// ---------------------------------------------------------------------------
// Generic WMMA bf16 GEMM:  C[M,N] = epi(A[M,K] @ B[K,N])
//   A: packed bf16 pairs [M][K/2] (ABF16) or f32 [M][K]
//   B: always pre-packed transposed Wt[N][K/2]
// Block tile 128x128, BK=32, 256 threads = 8 wave32 waves (2x4), each wave
// 4x2 16x16 wmma tiles. Ping-pong LDS; async global->LDS DMA when available.
// M always multiple of 128; BOUND guards N only.
// ---------------------------------------------------------------------------
enum { EPI_NONE = 0, EPI_TANH = 1, EPI_SILU = 2, EPI_DECAY = 3 };

template <int EPI, bool ABF16, bool BOUND>
__global__ __launch_bounds__(256) void gemm_wmma(
    const void* __restrict__ Av, const unsigned* __restrict__ Bt,
    float* __restrict__ Cout, int M, int N, int K,
    const float* __restrict__ bias) {
  __shared__ __align__(16) unsigned aU[2][128 * 20];  // rows stride 20 dwords
  __shared__ __align__(16) unsigned bU[2][128 * 20];  // (16B aligned, low-conflict)
  const int tid = threadIdx.x;
  const int m0 = blockIdx.y * 128;
  const int n0 = blockIdx.x * 128;
  const int w = tid >> 5, lane = tid & 31;
  const int wm = w >> 2, wn = w & 3;
  const int l16 = lane & 15, khalf = lane >> 4;
  const int K2 = K >> 1;
  const unsigned* Ap = (const unsigned*)Av;
  const float* Af = (const float*)Av;

  v8f acc[4][2];
#pragma unroll
  for (int i = 0; i < 4; ++i)
#pragma unroll
    for (int j = 0; j < 2; ++j)
#pragma unroll
      for (int e = 0; e < 8; ++e) acc[i][j][e] = 0.f;

  float4 raf[4];  // f32-A staging (conversion can't be DMA'd)

  auto loadA_regs = [&](int k0) {
#pragma unroll
    for (int it = 0; it < 4; ++it) {
      int e = it * 256 + tid;              // 0..1023 over [128 rows][8 quads]
      int r = e >> 3, c4 = e & 7;
      raf[it] = *(const float4*)(Af + (long)(m0 + r) * K + k0 + c4 * 4);
    }
  };
  auto storeA_regs = [&](int buf) {
#pragma unroll
    for (int it = 0; it < 4; ++it) {
      int e = it * 256 + tid;
      int r = e >> 3, c4 = e & 7;
      aU[buf][r * 20 + c4 * 2]     = pack_bf16(raf[it].x, raf[it].y);
      aU[buf][r * 20 + c4 * 2 + 1] = pack_bf16(raf[it].z, raf[it].w);
    }
  };

  const int KT = K >> 5;

#if USE_ASYNC_LDS
  // ---- async global->LDS DMA path (ASYNCcnt) ------------------------------
  auto issueTile = [&](int k0, int buf) {
    if (ABF16) {
#pragma unroll
      for (int it = 0; it < 2; ++it) {
        int e = it * 256 + tid;            // 0..511 over [128 rows][4 quads]
        int r = e >> 2, q4 = e & 3;
        __builtin_amdgcn_global_load_async_to_lds_b128(
            (GLOBAL_AS v4i*)(Ap + (long)(m0 + r) * K2 + (k0 >> 1) + q4 * 4),
            (LDS_AS v4i*)&aU[buf][r * 20 + q4 * 4], 0, 0);
      }
    }
#pragma unroll
    for (int it = 0; it < 2; ++it) {
      int e = it * 256 + tid;
      int n = e >> 2, q4 = e & 3;
      if (!BOUND || (n0 + n) < N)
        __builtin_amdgcn_global_load_async_to_lds_b128(
            (GLOBAL_AS v4i*)(Bt + (long)(n0 + n) * K2 + (k0 >> 1) + q4 * 4),
            (LDS_AS v4i*)&bU[buf][n * 20 + q4 * 4], 0, 0);
      else
        *(v4u*)&bU[buf][n * 20 + q4 * 4] = zero4();
    }
  };

  if (!ABF16) loadA_regs(0);
  issueTile(0, 0);
  for (int kt = 0; kt < KT; ++kt) {
    const int buf = kt & 1;
    if (!ABF16) {
      storeA_regs(buf);
      if (kt + 1 < KT) loadA_regs((kt + 1) << 5);
    }
    async_wait0();                  // this wave's DMA for tile kt complete
    __syncthreads();                // => whole tile complete, prev reads done
    if (kt + 1 < KT) issueTile((kt + 1) << 5, buf ^ 1);

    union Frag { v16bf v; v4u q[2]; };
    Frag fa[4], fb[2];
#pragma unroll
    for (int i = 0; i < 4; ++i) {
      const unsigned* base = &aU[buf][(wm * 64 + i * 16 + l16) * 20];
      fa[i].q[0] = *(const v4u*)(base + khalf * 4);
      fa[i].q[1] = *(const v4u*)(base + 8 + khalf * 4);
    }
#pragma unroll
    for (int j = 0; j < 2; ++j) {
      const unsigned* base = &bU[buf][(wn * 32 + j * 16 + l16) * 20];
      fb[j].q[0] = *(const v4u*)(base + khalf * 4);
      fb[j].q[1] = *(const v4u*)(base + 8 + khalf * 4);
    }
#pragma unroll
    for (int i = 0; i < 4; ++i)
#pragma unroll
      for (int j = 0; j < 2; ++j)
        acc[i][j] = __builtin_amdgcn_wmma_f32_16x16x32_bf16(
            false, fa[i].v, false, fb[j].v, (short)0, acc[i][j], false, false);
  }
#else
  // ---- register-staged fallback path --------------------------------------
  v4u ra[2], rb[2];
  auto loadA = [&](int k0) {
    if (ABF16) {
#pragma unroll
      for (int it = 0; it < 2; ++it) {
        int e = it * 256 + tid;
        int r = e >> 2, q4 = e & 3;
        ra[it] = *(const v4u*)(Ap + (long)(m0 + r) * K2 + (k0 >> 1) + q4 * 4);
      }
    } else {
      loadA_regs(k0);
    }
  };
  auto loadB = [&](int k0) {
#pragma unroll
    for (int it = 0; it < 2; ++it) {
      int e = it * 256 + tid;
      int n = e >> 2, q4 = e & 3;
      if (!BOUND || (n0 + n) < N)
        rb[it] = *(const v4u*)(Bt + (long)(n0 + n) * K2 + (k0 >> 1) + q4 * 4);
      else
        rb[it] = zero4();
    }
  };
  auto storeTiles = [&](int buf) {
    if (ABF16) {
#pragma unroll
      for (int it = 0; it < 2; ++it) {
        int e = it * 256 + tid;
        int r = e >> 2, q4 = e & 3;
        *(v4u*)&aU[buf][r * 20 + q4 * 4] = ra[it];
      }
    } else {
      storeA_regs(buf);
    }
#pragma unroll
    for (int it = 0; it < 2; ++it) {
      int e = it * 256 + tid;
      int n = e >> 2, q4 = e & 3;
      *(v4u*)&bU[buf][n * 20 + q4 * 4] = rb[it];
    }
  };

  loadA(0);
  loadB(0);
  for (int kt = 0; kt < KT; ++kt) {
    const int buf = kt & 1;
    storeTiles(buf);
    if (kt + 1 < KT) {
      loadA((kt + 1) << 5);
      loadB((kt + 1) << 5);
    }
    __syncthreads();

    union Frag { v16bf v; v4u q[2]; };
    Frag fa[4], fb[2];
#pragma unroll
    for (int i = 0; i < 4; ++i) {
      const unsigned* base = &aU[buf][(wm * 64 + i * 16 + l16) * 20];
      fa[i].q[0] = *(const v4u*)(base + khalf * 4);
      fa[i].q[1] = *(const v4u*)(base + 8 + khalf * 4);
    }
#pragma unroll
    for (int j = 0; j < 2; ++j) {
      const unsigned* base = &bU[buf][(wn * 32 + j * 16 + l16) * 20];
      fb[j].q[0] = *(const v4u*)(base + khalf * 4);
      fb[j].q[1] = *(const v4u*)(base + 8 + khalf * 4);
    }
#pragma unroll
    for (int i = 0; i < 4; ++i)
#pragma unroll
      for (int j = 0; j < 2; ++j)
        acc[i][j] = __builtin_amdgcn_wmma_f32_16x16x32_bf16(
            false, fa[i].v, false, fb[j].v, (short)0, acc[i][j], false, false);
  }
#endif

  // C/D layout: VGPR rr -> row (rr + 8*khalf), col = lane&15
#pragma unroll
  for (int i = 0; i < 4; ++i) {
    int growb = m0 + wm * 64 + i * 16 + khalf * 8;
#pragma unroll
    for (int j = 0; j < 2; ++j) {
      int gcol = n0 + wn * 32 + j * 16 + l16;
      if (!BOUND || gcol < N) {
        float bval = (EPI == EPI_DECAY) ? bias[gcol] : 0.f;
#pragma unroll
        for (int rr = 0; rr < 8; ++rr) {
          float val = acc[i][j][rr];
          if (EPI == EPI_TANH) val = tanhf(val);
          if (EPI == EPI_SILU) val = val / (1.f + __expf(-val));
          if (EPI == EPI_DECAY) val = __expf(-__expf(val + bval));
          Cout[(long)(growb + rr) * N + gcol] = val;
        }
      }
    }
  }
}

// ---------------------------------------------------------------------------
// mix (pair-per-thread): m[bt,5,32] einsum maa_w2[5,32,C], then the five
// token-shift interpolations, written directly as packed bf16 GEMM operands.
// ---------------------------------------------------------------------------
__global__ __launch_bounds__(256) void mix_kernel(
    const float* __restrict__ x, const float* __restrict__ xx,
    const float* __restrict__ m, const float* __restrict__ w2,
    const float* __restrict__ maa_w, const float* __restrict__ maa_k,
    const float* __restrict__ maa_v, const float* __restrict__ maa_r,
    const float* __restrict__ maa_g,
    unsigned* __restrict__ xw, unsigned* __restrict__ xk,
    unsigned* __restrict__ xv, unsigned* __restrict__ xr,
    unsigned* __restrict__ xg) {
  long gid = (long)blockIdx.x * 256 + threadIdx.x;   // over BT*C/2
  int c2 = (int)(gid % (Cc / 2));
  long bt = gid / (Cc / 2);
  int c = 2 * c2;
  const float* mrow = m + bt * 160;
  float s0[5], s1[5];
#pragma unroll
  for (int f = 0; f < 5; ++f) {
    float a0 = 0.f, a1 = 0.f;
#pragma unroll
    for (int d2 = 0; d2 < 32; ++d2) {
      float mv = mrow[f * 32 + d2];
      float2 wp = *(const float2*)(w2 + (long)(f * 32 + d2) * Cc + c);
      a0 += mv * wp.x;
      a1 += mv * wp.y;
    }
    s0[f] = a0;
    s1[f] = a1;
  }
  long base = bt * Cc + c;
  float2 xv2 = *(const float2*)(x + base);
  float2 xd2 = *(const float2*)(xx + base);
  float2 cw = *(const float2*)(maa_w + c);
  float2 ck = *(const float2*)(maa_k + c);
  float2 cv = *(const float2*)(maa_v + c);
  float2 cr = *(const float2*)(maa_r + c);
  float2 cg = *(const float2*)(maa_g + c);
  xw[gid] = pack_bf16(xv2.x + xd2.x * (cw.x + s0[0]), xv2.y + xd2.y * (cw.y + s1[0]));
  xk[gid] = pack_bf16(xv2.x + xd2.x * (ck.x + s0[1]), xv2.y + xd2.y * (ck.y + s1[1]));
  xv[gid] = pack_bf16(xv2.x + xd2.x * (cv.x + s0[2]), xv2.y + xd2.y * (cv.y + s1[2]));
  xr[gid] = pack_bf16(xv2.x + xd2.x * (cr.x + s0[3]), xv2.y + xd2.y * (cr.y + s1[3]));
  xg[gid] = pack_bf16(xv2.x + xd2.x * (cg.x + s0[4]), xv2.y + xd2.y * (cg.y + s1[4]));
}

// ---------------------------------------------------------------------------
// WKV6 scan: per (b,h); S[j][i] in registers, 4-way split over j.
// y_i = sum_j r_j*(S_ji + u_j*k_j*v_i);  S_ji = d_j*S_ji + k_j*v_i
// ---------------------------------------------------------------------------
__global__ __launch_bounds__(256) void wkv_kernel(
    const float* __restrict__ r, const float* __restrict__ k,
    const float* __restrict__ v, const float* __restrict__ dcy,
    const float* __restrict__ u, float* __restrict__ y) {
  const int b = blockIdx.x >> 5;
  const int h = blockIdx.x & 31;
  const int tid = threadIdx.x;
  const int i = tid & 63, q = tid >> 6;
  __shared__ float rs[64], ks[64], vs[64], ds[64], us[64], part[4][64];
  if (tid < 64) us[tid] = u[h * 64 + tid];
  float S[16];
#pragma unroll
  for (int jj = 0; jj < 16; ++jj) S[jj] = 0.f;
  __syncthreads();

  for (int t = 0; t < Tt; ++t) {
    long base = ((long)(b * Tt + t)) * Cc + h * 64;
    {
      const float* src = (q == 0) ? r : (q == 1) ? k : (q == 2) ? v : dcy;
      float* dst = (q == 0) ? rs : (q == 1) ? ks : (q == 2) ? vs : ds;
      dst[i] = src[base + i];
    }
    __syncthreads();
    float vi = vs[i];
    float yacc = 0.f;
#pragma unroll
    for (int jj = 0; jj < 16; ++jj) {
      int j = q * 16 + jj;
      float kv = ks[j] * vi;
      yacc += rs[j] * (S[jj] + us[j] * kv);
      S[jj] = ds[j] * S[jj] + kv;
    }
    part[q][i] = yacc;
    __syncthreads();
    if (q == 0) y[base + i] = part[0][i] + part[1][i] + part[2][i] + part[3][i];
  }
}

// ---------------------------------------------------------------------------
// GroupNorm over each head row of 64, then z = yn * g  (g = silu already)
// ---------------------------------------------------------------------------
__global__ __launch_bounds__(256) void gnorm_kernel(
    const float* __restrict__ y, const float* __restrict__ g,
    const float* __restrict__ lng, const float* __restrict__ lnb,
    float* __restrict__ z) {
  const int tid = threadIdx.x;
  long row = (long)blockIdx.x * 4 + (tid >> 6);
  const int i = tid & 63;
  long bt = row >> 5;
  int h = (int)(row & 31);
  long idx = bt * Cc + h * 64 + i;
  float val = y[idx];
  __shared__ float s1[256], s2[256];
  s1[tid] = val;
  s2[tid] = val * val;
  __syncthreads();
#pragma unroll
  for (int s = 32; s >= 1; s >>= 1) {
    if (i < s) { s1[tid] += s1[tid + s]; s2[tid] += s2[tid + s]; }
    __syncthreads();
  }
  int gb = tid & ~63;
  float mean = s1[gb] * (1.f / 64.f);
  float var = s2[gb] * (1.f / 64.f) - mean * mean;
  float yn = (val - mean) * rsqrtf(var + GN_EPS);
  yn = yn * lng[h * 64 + i] + lnb[h * 64 + i];
  z[idx] = yn * g[idx];
}

// ---------------------------------------------------------------------------
extern "C" void kernel_launch(void* const* d_in, const int* in_sizes, int n_in,
                              void* d_out, int out_size, void* d_ws,
                              size_t ws_size, hipStream_t stream) {
  const float* x       = (const float*)d_in[0];
  const float* maa_x   = (const float*)d_in[1];
  const float* maa_w   = (const float*)d_in[2];
  const float* maa_k   = (const float*)d_in[3];
  const float* maa_v   = (const float*)d_in[4];
  const float* maa_r   = (const float*)d_in[5];
  const float* maa_g   = (const float*)d_in[6];
  const float* tdecay  = (const float*)d_in[7];
  const float* u       = (const float*)d_in[8];
  const float* maa_w1  = (const float*)d_in[9];
  const float* maa_w2  = (const float*)d_in[10];
  const float* td_w1   = (const float*)d_in[11];
  const float* td_w2   = (const float*)d_in[12];
  const float* Wr      = (const float*)d_in[13];
  const float* Wk      = (const float*)d_in[14];
  const float* Wv      = (const float*)d_in[15];
  const float* Wg      = (const float*)d_in[16];
  const float* Wo      = (const float*)d_in[17];
  const float* ln_g    = (const float*)d_in[18];
  const float* ln_b    = (const float*)d_in[19];

  const long BTC = (long)BT * Cc;        // 4,194,304
  const long HBTC = BTC >> 1;            // packed dwords per [BT,C] tensor
  float* ws = (float*)d_ws;              // ~196 MB used
  float* xx   = ws + 0 * BTC;
  float* rbuf = ws + 1 * BTC;
  float* kbuf = ws + 2 * BTC;
  float* vbuf = ws + 3 * BTC;
  float* gbuf = ws + 4 * BTC;
  float* ybuf = ws + 5 * BTC;
  float* mbuf = ws + 6 * BTC;            // BT*160
  float* hbuf = mbuf + (long)BT * 160;   // BT*64
  unsigned* pk = (unsigned*)(hbuf + (long)BT * 64);
  unsigned* xxx_pk = pk + 0 * HBTC;
  unsigned* xw_pk  = pk + 1 * HBTC;
  unsigned* xk_pk  = pk + 2 * HBTC;
  unsigned* xv_pk  = pk + 3 * HBTC;
  unsigned* xr_pk  = pk + 4 * HBTC;
  unsigned* xg_pk  = pk + 5 * HBTC;
  unsigned* Wrt = pk + 6 * HBTC;         // [C][C/2] each
  unsigned* Wkt = pk + 7 * HBTC;
  unsigned* Wvt = pk + 8 * HBTC;
  unsigned* Wgt = pk + 9 * HBTC;
  unsigned* Wot = pk + 10 * HBTC;
  unsigned* w1t  = pk + 11 * HBTC;       // [160][1024]
  unsigned* tw1t = w1t + (long)160 * 1024;   // [64][1024]
  unsigned* tw2t = tw1t + (long)64 * 1024;   // [2048][32]
  // phase-safe aliases
  float* dbuf = xx;     // decay written after xx's last read (mix)
  float* zbuf = rbuf;   // z written after rbuf's last read (wkv)

  dim3 blk(256);
  dim3 gfull(16, 16);
  int nbp = (int)((HBTC + 255) / 256);   // pair-grain elementwise grids

  // one-time weight transpose-packs
  packT_kernel<<<dim3(32, 32), blk, 0, stream>>>(Wr, Wrt, Cc, Cc);
  packT_kernel<<<dim3(32, 32), blk, 0, stream>>>(Wk, Wkt, Cc, Cc);
  packT_kernel<<<dim3(32, 32), blk, 0, stream>>>(Wv, Wvt, Cc, Cc);
  packT_kernel<<<dim3(32, 32), blk, 0, stream>>>(Wg, Wgt, Cc, Cc);
  packT_kernel<<<dim3(32, 32), blk, 0, stream>>>(Wo, Wot, Cc, Cc);
  packT_kernel<<<dim3(3, 32), blk, 0, stream>>>(maa_w1, w1t, Cc, 160);
  packT_kernel<<<dim3(1, 32), blk, 0, stream>>>(td_w1, tw1t, Cc, 64);
  packT_kernel<<<dim3(32, 1), blk, 0, stream>>>(td_w2, tw2t, 64, Cc);

  prep_kernel<<<nbp, blk, 0, stream>>>(x, maa_x, xx, xxx_pk);
  gemm_wmma<EPI_TANH, true, true><<<dim3(2, 16), blk, 0, stream>>>(
      xxx_pk, w1t, mbuf, BT, 160, Cc, nullptr);
  mix_kernel<<<nbp, blk, 0, stream>>>(x, xx, mbuf, maa_w2, maa_w, maa_k, maa_v,
                                      maa_r, maa_g, xw_pk, xk_pk, xv_pk,
                                      xr_pk, xg_pk);
  gemm_wmma<EPI_TANH, true, true><<<dim3(1, 16), blk, 0, stream>>>(
      xw_pk, tw1t, hbuf, BT, 64, Cc, nullptr);
  gemm_wmma<EPI_DECAY, false, false><<<gfull, blk, 0, stream>>>(
      hbuf, tw2t, dbuf, BT, Cc, 64, tdecay);
  gemm_wmma<EPI_NONE, true, false><<<gfull, blk, 0, stream>>>(
      xr_pk, Wrt, rbuf, BT, Cc, Cc, nullptr);
  gemm_wmma<EPI_NONE, true, false><<<gfull, blk, 0, stream>>>(
      xk_pk, Wkt, kbuf, BT, Cc, Cc, nullptr);
  gemm_wmma<EPI_NONE, true, false><<<gfull, blk, 0, stream>>>(
      xv_pk, Wvt, vbuf, BT, Cc, Cc, nullptr);
  gemm_wmma<EPI_SILU, true, false><<<gfull, blk, 0, stream>>>(
      xg_pk, Wgt, gbuf, BT, Cc, Cc, nullptr);
  wkv_kernel<<<Bb * Hh, blk, 0, stream>>>(rbuf, kbuf, vbuf, dbuf, u, ybuf);
  gnorm_kernel<<<(BT * Hh) / 4, blk, 0, stream>>>(ybuf, gbuf, ln_g, ln_b, zbuf);
  gemm_wmma<EPI_NONE, false, false><<<gfull, blk, 0, stream>>>(
      zbuf, Wot, (float*)d_out, BT, Cc, Cc, nullptr);
}
